// FusedTriangleMultiplicativeUpdate_86371792322620
// MI455X (gfx1250) — compile-verified, matrix operand order checked by reading, not run on hardware
//
#include <hip/hip_runtime.h>
#include <hip/hip_bf16.h>

typedef __bf16 bf16;
typedef __attribute__((ext_vector_type(16))) __bf16 v16bf;
typedef __attribute__((ext_vector_type(8)))  __bf16 v8bf;
typedef __attribute__((ext_vector_type(4)))  __bf16 v4bf;
typedef __attribute__((ext_vector_type(8)))  float  v8f;

constexpr int kN  = 384;
constexpr int kN2 = kN * kN;        // 147456
constexpr int kCZ = 128;
constexpr int kCH = 128;
constexpr float kLnEps = 1e-5f;

// workspace byte offsets
constexpr size_t kWtOff = 0;                          // 768 rows x 128 bf16
constexpr size_t kAOff  = 256u * 1024u;
constexpr size_t kPlaneBytes = (size_t)kCH * kN2 * 2; // 37,748,736
constexpr size_t kBOff  = kAOff + kPlaneBytes;
constexpr size_t kXOff  = kBOff + kPlaneBytes;

// wT row bases (each row = 128 bf16 of K)
constexpr int kWtP  = 0;    // w_ab_p^T  : 256 rows
constexpr int kWtG  = 256;  // w_ab_g^T  : 256 rows
constexpr int kWtWG = 512;  // w_g^T     : 128 rows
constexpr int kWtWZ = 640;  // w_z^T     : 128 rows

__device__ __forceinline__ float sigm(float x) {
    return 1.0f / (1.0f + __expf(-x));
}

// Build a 16x32 bf16 A/B fragment per ISA layout:
// lanes 0-15 hold K = kbase+0..7 and kbase+16..23,
// lanes 16-31 hold K = kbase+8..15 and kbase+24..31 (lsel = 8*(lane>=16)).
__device__ __forceinline__ v16bf load_frag(const bf16* row, int kbase, int lsel) {
    v8bf lo = *(const v8bf*)(row + kbase + lsel);
    v8bf hi = *(const v8bf*)(row + kbase + lsel + 16);
    return __builtin_shufflevector(lo, hi, 0, 1, 2, 3, 4, 5, 6, 7,
                                           8, 9, 10, 11, 12, 13, 14, 15);
}

__device__ __forceinline__ v8f wmma_bf16(v16bf a, v16bf b, v8f c) {
    return __builtin_amdgcn_wmma_f32_16x16x32_bf16(
        /*neg_a=*/false, a, /*neg_b=*/false, b,
        /*c_mod=*/(short)0, c, /*reuse_a=*/false, /*reuse_b=*/false);
}

// Async memory->LDS copy (CDNA5 GLOBAL_LOAD_ASYNC_TO_LDS_B128, ASYNCcnt).
// lds_off: per-lane LDS byte offset (low 32 bits of flat LDS address),
// gaddr  : per-lane 64-bit global address. Copies 16 bytes per lane.
__device__ __forceinline__ void async_copy_b128(unsigned lds_off, const void* gptr) {
    unsigned long long ga = (unsigned long long)gptr;
    asm volatile("global_load_async_to_lds_b128 %0, %1, off"
                 :: "v"(lds_off), "v"(ga) : "memory");
}
__device__ __forceinline__ void wait_async0() {
    asm volatile("s_wait_asynccnt 0x0" ::: "memory");
}

// ---------------- Kernel 0: weights -> wT[out][k] bf16 ----------------
__global__ __launch_bounds__(256) void prep_weights(
    const float* __restrict__ w_ab_p, const float* __restrict__ w_ab_g,
    const float* __restrict__ w_g, const float* __restrict__ w_z,
    bf16* __restrict__ wT) {
    int idx = blockIdx.x * 256 + threadIdx.x;
    if (idx >= 768 * 128) return;
    int r = idx >> 7, k = idx & 127;
    float v;
    if (r < 256)      v = w_ab_p[k * 256 + r];
    else if (r < 512) v = w_ab_g[k * 256 + (r - 256)];
    else if (r < 640) v = w_g[k * 128 + (r - 512)];
    else              v = w_z[k * 128 + (r - 640)];
    wT[r * 128 + k] = (bf16)v;
}

// ---------------- Kernel 1: LN(z) -> (p,gate) GEMMs -> a,b planes ----------------
__global__ __launch_bounds__(256) void k1_proj(
    const float* __restrict__ z, const float* __restrict__ mask,
    const float* __restrict__ lnw, const float* __restrict__ lnb,
    const bf16* __restrict__ wT,
    bf16* __restrict__ aB, bf16* __restrict__ bB) {
    __shared__ bf16 zt[8][16][136];  // per-wave LN'd tile, padded rows
    int tid = threadIdx.x, wv = tid >> 5, lane = tid & 31;
    int m_base = (blockIdx.x * 8 + wv) * 16;

    // ---- LayerNorm: 2 lanes per row, 64 floats each ----
    int row = lane >> 1, half = lane & 1;
    const float* zr = z + (size_t)(m_base + row) * kCZ + half * 64;
    float4 t[16];
    float s = 0.f, s2 = 0.f;
#pragma unroll
    for (int q = 0; q < 16; q++) {
        t[q] = *(const float4*)(zr + q * 4);
        s  += t[q].x + t[q].y + t[q].z + t[q].w;
        s2 += t[q].x * t[q].x + t[q].y * t[q].y + t[q].z * t[q].z + t[q].w * t[q].w;
    }
    s  += __shfl_xor(s, 1, 32);
    s2 += __shfl_xor(s2, 1, 32);
    float mean = s * (1.f / 128.f);
    float var  = s2 * (1.f / 128.f) - mean * mean;
    float rstd = rsqrtf(var + kLnEps);
#pragma unroll
    for (int q = 0; q < 16; q++) {
        float4 wq = *(const float4*)(lnw + half * 64 + q * 4);
        float4 bq = *(const float4*)(lnb + half * 64 + q * 4);
        v4bf o;
        o[0] = (bf16)((t[q].x - mean) * rstd * wq.x + bq.x);
        o[1] = (bf16)((t[q].y - mean) * rstd * wq.y + bq.y);
        o[2] = (bf16)((t[q].z - mean) * rstd * wq.z + bq.z);
        o[3] = (bf16)((t[q].w - mean) * rstd * wq.w + bq.w);
        *(v4bf*)&zt[wv][row][half * 64 + q * 4] = o;
    }

    // ---- WMMA: 16 col-tiles over 256 channels; p and gate per tile ----
    int col = lane & 15, hi = lane >> 4, lsel = hi * 8;
    float mk[8];
#pragma unroll
    for (int r = 0; r < 8; r++) mk[r] = mask[m_base + r + 8 * hi];
    const bf16* arow = &zt[wv][col][0];

    for (int ct = 0; ct < 16; ct++) {
        v8f accp = {}, accg = {};
        const bf16* bp = wT + (size_t)(kWtP + ct * 16 + col) * 128;
        const bf16* bg = wT + (size_t)(kWtG + ct * 16 + col) * 128;
#pragma unroll
        for (int kk = 0; kk < 4; kk++) {
            v16bf af  = load_frag(arow, kk * 32, lsel);
            v16bf bfp = load_frag(bp,   kk * 32, lsel);
            v16bf bfg = load_frag(bg,   kk * 32, lsel);
            accp = wmma_bf16(af, bfp, accp);
            accg = wmma_bf16(af, bfg, accg);
        }
        int ch = ct * 16 + col;                       // interleaved channel
        bf16* dst = (ch & 1) ? bB : aB;               // even -> a, odd -> b
        size_t coff = (size_t)(ch >> 1) * kN2;
#pragma unroll
        for (int r = 0; r < 8; r++) {
            int m = m_base + r + 8 * hi;
            float v = accp[r] * mk[r] * sigm(accg[r]);
            dst[coff + m] = (bf16)v;
        }
    }
}

// ---------------- Kernel 2: per-channel X = A * B^T (batched 384^3 GEMM) ----
// 128x64 block tile, 8 waves, 2x2 subtiles per wave, double-buffered LDS
// filled by CDNA5 async-to-LDS copies (ASYNCcnt).
__global__ __launch_bounds__(256) void k2_einsum(
    const bf16* __restrict__ aB, const bf16* __restrict__ bB,
    bf16* __restrict__ xB) {
    __shared__ bf16 At[2][128][40];   // stride 40 elems = 80B -> conflict-free frags
    __shared__ bf16 Bt[2][64][40];
    int tid = threadIdx.x, w = tid >> 5, lane = tid & 31;
    int i0 = blockIdx.x * 128, j0 = blockIdx.y * 64, c = blockIdx.z;
    size_t cb = (size_t)c * kN2;

    // async staging assignment: A = 512 chunks of 8 bf16 (2/thread), B = 256 (1/thread)
    int qa = tid * 2;
    int ar0 = qa >> 2,      ac0 = (qa & 3) * 8;
    int ar1 = (qa + 1) >> 2, ac1 = ((qa + 1) & 3) * 8;
    int br  = tid >> 2,     bc  = (tid & 3) * 8;
    const bf16* ga0 = aB + cb + (size_t)(i0 + ar0) * kN + ac0;
    const bf16* ga1 = aB + cb + (size_t)(i0 + ar1) * kN + ac1;
    const bf16* gb  = bB + cb + (size_t)(j0 + br) * kN + bc;

    // wave compute mapping: row subtiles {ri,ri+1} of 8, col subtiles {ci,ci+1} of 4
    int ri = (w & 3) * 2, ci = (w >> 2) * 2;
    int col = lane & 15, hi = lane >> 4, lsel = hi * 8;
    v8f acc00 = {}, acc01 = {}, acc10 = {}, acc11 = {};

    auto issue = [&](int buf, int k0) {
        async_copy_b128((unsigned)(size_t)&At[buf][ar0][ac0], ga0 + k0);
        async_copy_b128((unsigned)(size_t)&At[buf][ar1][ac1], ga1 + k0);
        async_copy_b128((unsigned)(size_t)&Bt[buf][br][bc],   gb  + k0);
    };

    issue(0, 0);
    for (int it = 0; it < kN / 32; ++it) {
        wait_async0();
        __syncthreads();                       // stage it&1 fully in LDS
        if (it + 1 < kN / 32) issue((it + 1) & 1, (it + 1) * 32);
        int buf = it & 1;
        v16bf a0 = load_frag(&At[buf][ri * 16 + col][0], 0, lsel);
        v16bf a1 = load_frag(&At[buf][(ri + 1) * 16 + col][0], 0, lsel);
        v16bf b0 = load_frag(&Bt[buf][ci * 16 + col][0], 0, lsel);
        v16bf b1 = load_frag(&Bt[buf][(ci + 1) * 16 + col][0], 0, lsel);
        acc00 = wmma_bf16(a0, b0, acc00);
        acc01 = wmma_bf16(a0, b1, acc01);
        acc10 = wmma_bf16(a1, b0, acc10);
        acc11 = wmma_bf16(a1, b1, acc11);
    }
#pragma unroll
    for (int r = 0; r < 8; r++) {
        int r0 = i0 + ri * 16 + r + 8 * hi;
        int r1 = r0 + 16;
        int c0 = j0 + ci * 16 + col;
        xB[cb + (size_t)r0 * kN + c0]      = (bf16)acc00[r];
        xB[cb + (size_t)r0 * kN + c0 + 16] = (bf16)acc01[r];
        xB[cb + (size_t)r1 * kN + c0]      = (bf16)acc10[r];
        xB[cb + (size_t)r1 * kN + c0 + 16] = (bf16)acc11[r];
    }
}

// ---------------- Kernel 3: LN(x)@w_z * sigmoid(LN(z)@w_g) ----------------
__global__ __launch_bounds__(128) void k3_out(
    const float* __restrict__ z,
    const float* __restrict__ ln_in_w, const float* __restrict__ ln_in_b,
    const float* __restrict__ ln_out_w, const float* __restrict__ ln_out_b,
    const bf16* __restrict__ wT, const bf16* __restrict__ xB,
    float* __restrict__ out) {
    __shared__ bf16 zt[4][16][136];
    __shared__ bf16 xt[4][16][136];
    int tid = threadIdx.x, wv = tid >> 5, lane = tid & 31;
    int m_base = (blockIdx.x * 4 + wv) * 16;
    int row = lane >> 1, half = lane & 1;

    // ---- z LayerNorm (two-pass; second pass is L2-hot) ----
    {
        const float* zr = z + (size_t)(m_base + row) * kCZ + half * 64;
        float s = 0.f, s2 = 0.f;
#pragma unroll
        for (int q = 0; q < 16; q++) {
            float4 t = *(const float4*)(zr + q * 4);
            s  += t.x + t.y + t.z + t.w;
            s2 += t.x * t.x + t.y * t.y + t.z * t.z + t.w * t.w;
        }
        s  += __shfl_xor(s, 1, 32);
        s2 += __shfl_xor(s2, 1, 32);
        float mean = s * (1.f / 128.f);
        float rstd = rsqrtf(s2 * (1.f / 128.f) - mean * mean + kLnEps);
#pragma unroll
        for (int q = 0; q < 16; q++) {
            float4 t  = *(const float4*)(zr + q * 4);
            float4 wq = *(const float4*)(ln_in_w + half * 64 + q * 4);
            float4 bq = *(const float4*)(ln_in_b + half * 64 + q * 4);
            v4bf o;
            o[0] = (bf16)((t.x - mean) * rstd * wq.x + bq.x);
            o[1] = (bf16)((t.y - mean) * rstd * wq.y + bq.y);
            o[2] = (bf16)((t.z - mean) * rstd * wq.z + bq.z);
            o[3] = (bf16)((t.w - mean) * rstd * wq.w + bq.w);
            *(v4bf*)&zt[wv][row][half * 64 + q * 4] = o;
        }
    }

    // ---- x LayerNorm over channels (x stored [c][m], strided reads) ----
    {
        size_t m = (size_t)m_base + row;
        float s = 0.f, s2 = 0.f;
        for (int q = 0; q < 64; q++) {
            int cc = half * 64 + q;
            float v = (float)xB[(size_t)cc * kN2 + m];
            s += v; s2 += v * v;
        }
        s  += __shfl_xor(s, 1, 32);
        s2 += __shfl_xor(s2, 1, 32);
        float mean = s * (1.f / 128.f);
        float rstd = rsqrtf(s2 * (1.f / 128.f) - mean * mean + kLnEps);
        for (int q = 0; q < 64; q++) {
            int cc = half * 64 + q;
            float v = (float)xB[(size_t)cc * kN2 + m];
            xt[wv][row][cc] = (bf16)((v - mean) * rstd * ln_out_w[cc] + ln_out_b[cc]);
        }
    }

    // ---- WMMA: x@w_z and z_ln@w_g, fuse gate ----
    int col = lane & 15, hi = lane >> 4, lsel = hi * 8;
    const bf16* ax = &xt[wv][col][0];
    const bf16* az = &zt[wv][col][0];
    for (int ct = 0; ct < 8; ct++) {
        v8f accx = {}, accg = {};
        const bf16* bz = wT + (size_t)(kWtWZ + ct * 16 + col) * 128;
        const bf16* bg = wT + (size_t)(kWtWG + ct * 16 + col) * 128;
#pragma unroll
        for (int kk = 0; kk < 4; kk++) {
            accx = wmma_bf16(load_frag(ax, kk * 32, lsel),
                             load_frag(bz, kk * 32, lsel), accx);
            accg = wmma_bf16(load_frag(az, kk * 32, lsel),
                             load_frag(bg, kk * 32, lsel), accg);
        }
#pragma unroll
        for (int r = 0; r < 8; r++) {
            int m = m_base + r + 8 * hi;
            out[(size_t)m * kCZ + ct * 16 + col] = accx[r] * sigm(accg[r]);
        }
    }
}

extern "C" void kernel_launch(void* const* d_in, const int* in_sizes, int n_in,
                              void* d_out, int out_size, void* d_ws, size_t ws_size,
                              hipStream_t stream) {
    const float* z        = (const float*)d_in[0];
    const float* mask     = (const float*)d_in[1];
    const float* ln_in_w  = (const float*)d_in[2];
    const float* ln_in_b  = (const float*)d_in[3];
    const float* w_ab_p   = (const float*)d_in[4];
    const float* w_ab_g   = (const float*)d_in[5];
    const float* w_g      = (const float*)d_in[6];
    const float* ln_out_w = (const float*)d_in[7];
    const float* ln_out_b = (const float*)d_in[8];
    const float* w_z      = (const float*)d_in[9];
    (void)in_sizes; (void)n_in; (void)out_size; (void)ws_size;

    char* ws  = (char*)d_ws;
    bf16* wT  = (bf16*)(ws + kWtOff);
    bf16* aB  = (bf16*)(ws + kAOff);
    bf16* bB  = (bf16*)(ws + kBOff);
    bf16* xB  = (bf16*)(ws + kXOff);

    prep_weights<<<384, 256, 0, stream>>>(w_ab_p, w_ab_g, w_g, w_z, wT);
    k1_proj<<<kN2 / 128, 256, 0, stream>>>(z, mask, ln_in_w, ln_in_b, wT, aB, bB);
    k2_einsum<<<dim3(kN / 128, kN / 64, kCH), 256, 0, stream>>>(aB, bB, xB);
    k3_out<<<kN2 / 64, 128, 0, stream>>>(z, ln_in_w, ln_in_b, ln_out_w, ln_out_b,
                                         wT, xB, (float*)d_out);
}
